// Encoder_59708635349032
// MI455X (gfx1250) — matrix operand order repaired and checked by reading, not compiled, not run on hardware
//
#include <hip/hip_runtime.h>

// ---------------------------------------------------------------------------
// Problem constants (from the reference): B=16, T=64, D=256, H=1536
// ---------------------------------------------------------------------------
#define BB    16
#define TT    64
#define DIN   256
#define HH    1536
#define FOURH 6144          // 4*H
#define STEPS 1008          // (T-1)*B
#define NWG   48            // persistent scan workgroups: 48 * 8 waves = 384 = 6144/16
#define WAVES_PER_BLK 8

typedef __bf16 v16bf __attribute__((ext_vector_type(16)));
typedef __bf16 v8bf  __attribute__((ext_vector_type(8)));
typedef float  v8f   __attribute__((ext_vector_type(8)));

// ---------------------------------------------------------------------------
// helpers
// ---------------------------------------------------------------------------
__device__ __host__ inline __bf16 f2bf(float x) {
    union { unsigned u; float f; } v; v.f = x;
    unsigned r = v.u + 0x7FFFu + ((v.u >> 16) & 1u);   // round-to-nearest-even
    union { unsigned short s; __bf16 h; } o; o.s = (unsigned short)(r >> 16);
    return o.h;
}

__device__ inline float sigmoidf_(float x) { return 1.0f / (1.0f + __expf(-x)); }

// A-matrix 16x32 bf16 tile: lane m=lane&15, K-base = (lane>=16)*16,
// elements 0..15 -> K=kb..kb+15 (contiguous, matches B loaders).
__device__ inline v16bf load_a(const __bf16* __restrict__ W, int ld, int j0, int k0, int lane) {
    const int m  = lane & 15;
    const int kb = (lane >> 4) << 4;
    const __bf16* p = W + (size_t)(j0 + m) * (size_t)ld + (size_t)(k0 + kb);
    v8bf lo = *(const v8bf*)p;          // 16B load
    v8bf hi = *(const v8bf*)(p + 8);    // 16B load
    return __builtin_shufflevector(lo, hi, 0,1,2,3,4,5,6,7,8,9,10,11,12,13,14,15);
}

// B-matrix 32x16 bf16: lane n=lane&15, K = (lane>=16)*16 + e (16 contiguous).
// GEMV case: B[k][n] = h[k] for every n -> broadcast the vector chunk.
__device__ inline v16bf load_hvec(const __bf16* __restrict__ h, int k0, int lane) {
    const int kb = (lane >> 4) << 4;
    return *(const v16bf*)(h + k0 + kb);   // 32B load
}
// GEMM case: B[k][n] = W[j0+n][k] (W row-major [out, in], so W^T is K x N).
__device__ inline v16bf load_bw(const __bf16* __restrict__ W, int ld, int j0, int k0, int lane) {
    const int n  = lane & 15;
    const int kb = (lane >> 4) << 4;
    return *(const v16bf*)(W + (size_t)(j0 + n) * (size_t)ld + (size_t)(k0 + kb));
}

__device__ inline float vpick8(v8f v, int r) {
    float x = v[0];
#pragma unroll
    for (int i = 1; i < 8; ++i) if (r == i) x = v[i];
    return x;
}

// Device-wide barrier: monotonically increasing counter, no reset races.
__device__ inline void grid_barrier(unsigned* cnt, unsigned target) {
    __syncthreads();
    if (threadIdx.x == 0) {
        __threadfence();                       // release prior writes
        atomicAdd(cnt, 1u);
        while (__hip_atomic_load(cnt, __ATOMIC_RELAXED, __HIP_MEMORY_SCOPE_AGENT) < target)
            __builtin_amdgcn_s_sleep(1);
    }
    __syncthreads();
    __threadfence();                           // acquire: invalidate L0 so fresh data is seen
}

// ---------------------------------------------------------------------------
// Kernel 1: convert all four weight matrices to bf16 (L2-resident working set)
// ---------------------------------------------------------------------------
__global__ void k_convert(const float* __restrict__ wih0, const float* __restrict__ whh0,
                          const float* __restrict__ wih1, const float* __restrict__ whh1,
                          __bf16* __restrict__ wih0b, __bf16* __restrict__ whh0b,
                          __bf16* __restrict__ wih1b, __bf16* __restrict__ whh1b) {
    const size_t big   = (size_t)FOURH * HH;   // 9,437,184
    const size_t small = (size_t)FOURH * DIN;  // 1,572,864
    size_t i      = (size_t)blockIdx.x * blockDim.x + threadIdx.x;
    size_t stride = (size_t)gridDim.x * blockDim.x;
    for (size_t idx = i; idx < big; idx += stride) {
        whh0b[idx] = f2bf(whh0[idx]);
        wih1b[idx] = f2bf(wih1[idx]);
        whh1b[idx] = f2bf(whh1[idx]);
        if (idx < small) wih0b[idx] = f2bf(wih0[idx]);
    }
}

// ---------------------------------------------------------------------------
// Kernel 2: build bf16 seq (ei-major, b-minor, t=ei+1), fused biases, zero state
// ---------------------------------------------------------------------------
__global__ void k_prep(const float* __restrict__ batch,
                       const float* __restrict__ bi0, const float* __restrict__ bh0,
                       const float* __restrict__ bi1, const float* __restrict__ bh1,
                       __bf16* __restrict__ seqb, float* __restrict__ b0, float* __restrict__ b1,
                       float* __restrict__ h0, float* __restrict__ c0,
                       float* __restrict__ h1, float* __restrict__ c1,
                       __bf16* __restrict__ h0b, __bf16* __restrict__ h1b,
                       unsigned* __restrict__ bar) {
    int idx    = blockIdx.x * blockDim.x + threadIdx.x;
    int stride = gridDim.x * blockDim.x;
    for (int i = idx; i < STEPS * DIN; i += stride) {
        int s = i >> 8, d = i & (DIN - 1);
        int ei = s >> 4, b = s & 15;                        // s = ei*16 + b
        seqb[i] = f2bf(batch[((size_t)b * TT + (ei + 1)) * DIN + d]);
    }
    for (int j = idx; j < FOURH; j += stride) { b0[j] = bi0[j] + bh0[j]; b1[j] = bi1[j] + bh1[j]; }
    for (int t = idx; t < HH; t += stride) {
        h0[t] = 0.f; c0[t] = 0.f; h1[t] = 0.f; c1[t] = 0.f;
        h0b[t] = f2bf(0.f); h1b[t] = f2bf(0.f);
    }
    if (idx == 0) { bar[0] = 0u; bar[1] = 0u; }
}

// ---------------------------------------------------------------------------
// Kernel 3: x_pre0 = seq @ W_ih0^T + b0   (1008 x 6144, K=256) — dense WMMA GEMM
// one 16x16 output tile per wave; 8 x v_wmma_f32_16x16x32_bf16 per tile
// ---------------------------------------------------------------------------
__global__ void __launch_bounds__(256) k_gemm(const __bf16* __restrict__ seqb,
                                              const __bf16* __restrict__ wih0b,
                                              const float* __restrict__ b0,
                                              float* __restrict__ xpre0) {
    const int lane  = threadIdx.x & 31;
    const int gwave = blockIdx.x * WAVES_PER_BLK + (threadIdx.x >> 5);  // 0 .. 24191
    const int ti = gwave / 384;
    const int tj = gwave - ti * 384;
    const int s0 = ti << 4, j0 = tj << 4;
    v8f acc = {0.f,0.f,0.f,0.f,0.f,0.f,0.f,0.f};
#pragma unroll
    for (int k0 = 0; k0 < DIN; k0 += 32) {
        v16bf a = load_a (seqb,  DIN, s0, k0, lane);
        v16bf b = load_bw(wih0b, DIN, j0, k0, lane);
        acc = __builtin_amdgcn_wmma_f32_16x16x32_bf16(false, a, false, b, (short)0, acc, false, false);
    }
    const int n = lane & 15, half = lane >> 4;
    const float bias = b0[j0 + n];
#pragma unroll
    for (int r = 0; r < 8; ++r) {
        int row = s0 + r + (half << 3);
        xpre0[(size_t)row * FOURH + j0 + n] = acc[r] + bias;
    }
}

// ---------------------------------------------------------------------------
// Kernel 4: persistent sequential scan. 384 waves, each owns one 16-row slice
// of the 6144 gate pre-activations; full K=1536 reduction = 48 WMMAs/matrix.
// 4 grid barriers per step. Weights stay hot in the 192MB L2 (56.6 MB bf16).
// ---------------------------------------------------------------------------
__global__ void __launch_bounds__(256, 1) k_scan(
        const __bf16* __restrict__ whh0, const __bf16* __restrict__ wih1,
        const __bf16* __restrict__ whh1, const float* __restrict__ xpre0,
        const float* __restrict__ b1,
        float* __restrict__ pre0, float* __restrict__ pre1,
        float* __restrict__ h0, float* __restrict__ c0,
        float* __restrict__ h1, float* __restrict__ c1,
        __bf16* __restrict__ h0b, __bf16* __restrict__ h1b,
        unsigned* __restrict__ bar, float* __restrict__ out) {
    const int lane  = threadIdx.x & 31;
    const int gwave = blockIdx.x * WAVES_PER_BLK + (threadIdx.x >> 5);  // 0..383
    const int j0    = gwave << 4;
    const int gt    = blockIdx.x * 256 + threadIdx.x;
    const int half  = lane >> 4;
    const int ncol  = lane & 15;
    const int rsel  = lane & 7;
    const bool wlane = ncol < 8;                  // 16 writer lanes per wave
    const int mrow  = rsel + (half << 3);
    unsigned round = 0;

    for (int s = 0; s < STEPS; ++s) {
        // ---- layer 0: pre0 = x_pre0[s] + W_hh0 @ h0 ----
        v8f acc = {0.f,0.f,0.f,0.f,0.f,0.f,0.f,0.f};
        for (int k0 = 0; k0 < HH; k0 += 32) {
            v16bf a = load_a(whh0, HH, j0, k0, lane);
            v16bf b = load_hvec(h0b, k0, lane);
            acc = __builtin_amdgcn_wmma_f32_16x16x32_bf16(false, a, false, b, (short)0, acc, false, false);
        }
        if (wlane) {
            int j = j0 + mrow;
            pre0[j] = vpick8(acc, rsel) + xpre0[(size_t)s * FOURH + j];
        }
        ++round; grid_barrier(bar, round * NWG);

        // ---- layer-0 cell ----
        if (gt < HH) {
            float iv = pre0[gt], fv = pre0[gt + HH], gv = pre0[gt + 2*HH], ov = pre0[gt + 3*HH];
            float cn = sigmoidf_(fv) * c0[gt] + sigmoidf_(iv) * tanhf(gv);
            float hn = sigmoidf_(ov) * tanhf(cn);
            c0[gt] = cn; h0[gt] = hn; h0b[gt] = f2bf(hn);
        }
        ++round; grid_barrier(bar, round * NWG);

        // ---- layer 1: pre1 = W_ih1 @ h0 + W_hh1 @ h1 + b1 ----
        acc = (v8f){0.f,0.f,0.f,0.f,0.f,0.f,0.f,0.f};
        for (int k0 = 0; k0 < HH; k0 += 32) {
            v16bf a = load_a(wih1, HH, j0, k0, lane);
            v16bf b = load_hvec(h0b, k0, lane);
            acc = __builtin_amdgcn_wmma_f32_16x16x32_bf16(false, a, false, b, (short)0, acc, false, false);
        }
        for (int k0 = 0; k0 < HH; k0 += 32) {
            v16bf a = load_a(whh1, HH, j0, k0, lane);
            v16bf b = load_hvec(h1b, k0, lane);
            acc = __builtin_amdgcn_wmma_f32_16x16x32_bf16(false, a, false, b, (short)0, acc, false, false);
        }
        if (wlane) {
            int j = j0 + mrow;
            pre1[j] = vpick8(acc, rsel) + b1[j];
        }
        ++round; grid_barrier(bar, round * NWG);

        // ---- layer-1 cell ----
        if (gt < HH) {
            float iv = pre1[gt], fv = pre1[gt + HH], gv = pre1[gt + 2*HH], ov = pre1[gt + 3*HH];
            float cn = sigmoidf_(fv) * c1[gt] + sigmoidf_(iv) * tanhf(gv);
            float hn = sigmoidf_(ov) * tanhf(cn);
            c1[gt] = cn; h1[gt] = hn; h1b[gt] = f2bf(hn);
        }
        ++round; grid_barrier(bar, round * NWG);
    }

    // d_out = [h_final(2,H), c_final(2,H)] flat
    if (gt < HH) {
        out[gt]          = h0[gt];
        out[HH + gt]     = h1[gt];
        out[2*HH + gt]   = c0[gt];
        out[3*HH + gt]   = c1[gt];
    }
}

// ---------------------------------------------------------------------------
// host launcher
// ---------------------------------------------------------------------------
extern "C" void kernel_launch(void* const* d_in, const int* in_sizes, int n_in,
                              void* d_out, int out_size, void* d_ws, size_t ws_size,
                              hipStream_t stream) {
    (void)in_sizes; (void)n_in; (void)out_size; (void)ws_size;
    const float* batch = (const float*)d_in[0];
    const float* Wih0  = (const float*)d_in[1];
    const float* Whh0  = (const float*)d_in[2];
    const float* bih0  = (const float*)d_in[3];
    const float* bhh0  = (const float*)d_in[4];
    const float* Wih1  = (const float*)d_in[5];
    const float* Whh1  = (const float*)d_in[6];
    const float* bih1  = (const float*)d_in[7];
    const float* bhh1  = (const float*)d_in[8];
    float* out = (float*)d_out;

    char* w = (char*)d_ws;
    size_t o = 0;
    auto take = [&](size_t bytes) -> char* {
        char* p = w + o;
        o = (o + bytes + 255) & ~(size_t)255;
        return p;
    };
    unsigned* bar   = (unsigned*)take(256);
    float*    xpre0 = (float*)   take((size_t)STEPS * FOURH * 4);   // 24.8 MB
    __bf16*   wih0b = (__bf16*)  take((size_t)FOURH * DIN * 2);     //  3.1 MB
    __bf16*   whh0b = (__bf16*)  take((size_t)FOURH * HH  * 2);     // 18.9 MB
    __bf16*   wih1b = (__bf16*)  take((size_t)FOURH * HH  * 2);     // 18.9 MB
    __bf16*   whh1b = (__bf16*)  take((size_t)FOURH * HH  * 2);     // 18.9 MB
    __bf16*   seqb  = (__bf16*)  take((size_t)STEPS * DIN * 2);     //  0.5 MB
    float*    b0    = (float*)   take(FOURH * 4);
    float*    b1    = (float*)   take(FOURH * 4);
    float*    pre0  = (float*)   take(FOURH * 4);
    float*    pre1  = (float*)   take(FOURH * 4);
    float*    h0    = (float*)   take(HH * 4);
    float*    c0    = (float*)   take(HH * 4);
    float*    h1    = (float*)   take(HH * 4);
    float*    c1    = (float*)   take(HH * 4);
    __bf16*   h0b   = (__bf16*)  take(HH * 2);
    __bf16*   h1b   = (__bf16*)  take(HH * 2);

    k_convert<<<2048, 256, 0, stream>>>(Wih0, Whh0, Wih1, Whh1, wih0b, whh0b, wih1b, whh1b);
    k_prep<<<1008, 256, 0, stream>>>(batch, bih0, bhh0, bih1, bhh1, seqb, b0, b1,
                                     h0, c0, h1, c1, h0b, h1b, bar);
    // 63*384 = 24192 tiles, 8 waves/block -> 3024 blocks
    k_gemm<<<3024, 256, 0, stream>>>(seqb, wih0b, b0, xpre0);
    k_scan<<<NWG, 256, 0, stream>>>(whh0b, wih1b, whh1b, xpre0, b1,
                                    pre0, pre1, h0, c0, h1, c1, h0b, h1b, bar, out);
}
